// C2fDCNAttn_79070347919450
// MI455X (gfx1250) — matrix operand order repaired and checked by reading, not compiled, not run on hardware
//
#include <hip/hip_runtime.h>

// ---------------------------------------------------------------------------
// C2fDCNAttn for MI455X (gfx1250, wave32).
// bf16 activations/weights everywhere (packed once per launch), fp32 WMMA
// accumulation via v_wmma_f32_16x16x32_bf16. GEMM tiles staged with
// global_load_async_to_lds_b128 + s_wait_asynccnt, double-buffered LDS.
// Shapes: B=4, H=W=64 (M=16384 pixels), C=128, C1=256, C2=256, NT=80, GCD=512.
// ---------------------------------------------------------------------------

typedef __bf16 bf16;
typedef bf16  v16bf __attribute__((ext_vector_type(16)));
typedef bf16  v8bf  __attribute__((ext_vector_type(8)));
typedef float v8f   __attribute__((ext_vector_type(8)));

union Frag { v16bf v; v8bf h[2]; };

#define LDS_STRIDE 48  // 48 bf16 = 96B rows: 16B-aligned b128 fragment reads

__device__ __forceinline__ float silu_f(float v) { return v / (1.f + expf(-v)); }

// --- CDNA5 async global->LDS copy (ASYNCcnt-tracked), see CDNA5 ISA
__device__ __forceinline__ void async_cp16(const void* gaddr, unsigned lds_off) {
  asm volatile("global_load_async_to_lds_b128 %0, %1, off"
               :: "v"(lds_off), "v"(gaddr) : "memory");
}
__device__ __forceinline__ void async_wait0() {
  asm volatile("s_wait_asynccnt 0x0" ::: "memory");
}
__device__ __forceinline__ unsigned lds_off_of(const void* p) {
  return (unsigned)(unsigned long long)p;  // low 32 bits of flat LDS address
}

// ---------------------------------------------------------------------------
// Weight / activation packing kernels (run once per launch; tiny).
// ---------------------------------------------------------------------------
__global__ void pack_copy(const float* __restrict__ in, bf16* __restrict__ out, int n) {
  const int i = blockIdx.x * blockDim.x + threadIdx.x;
  if (i < n) out[i] = (bf16)in[i];
}
// [K,N] f32 -> [N,K] bf16
__global__ void pack_t(const float* __restrict__ in, bf16* __restrict__ out, int K, int N) {
  const int i = blockIdx.x * blockDim.x + threadIdx.x;
  if (i < K * N) {
    const int n = i / K, k = i - n * K;
    out[i] = (bf16)in[(size_t)k * N + n];
  }
}
// OIHW [N,K,3,3] f32 -> [9][N][K] bf16
__global__ void pack_conv(const float* __restrict__ in, bf16* __restrict__ out, int K, int N) {
  const int i = blockIdx.x * blockDim.x + threadIdx.x;
  if (i < 9 * N * K) {
    const int tap = i / (N * K), r = i - tap * N * K;
    const int n = r / K, k = r - n * K;
    out[i] = (bf16)in[((size_t)n * K + k) * 9 + tap];
  }
}
// NCHW f32 [B,Cc,4096] -> channels-last bf16 [B*4096, Cc]
__global__ void pack_nchw(const float* __restrict__ x, bf16* __restrict__ out, int Cc) {
  const size_t i = (size_t)blockIdx.x * blockDim.x + threadIdx.x;
  const int p = (int)(i & 4095);
  const int c = (int)((i >> 12) % Cc);
  const int b = (int)(i / ((size_t)Cc * 4096));
  out[((size_t)b * 4096 + p) * Cc + c] = (bf16)x[i];
}

// ---------------------------------------------------------------------------
// WMMA GEMM, bf16 in / bf16-or-f32 out, async double-buffered LDS staging.
// A: channels-last bf16 [M, ldA]; Wp: packed bf16 [N, K].
// 128 threads (4 waves), 64x64 tile, 32x32 per wave, K-step 32.
// ---------------------------------------------------------------------------
template <bool OUT_NCHW, bool HAS_BIAS, bool HAS_GB, bool SILU>
__global__ void gemm_wmma(const bf16* __restrict__ A, int ldA,
                          const bf16* __restrict__ Wp, int K,
                          const float* __restrict__ bias,
                          const float* __restrict__ gamma,
                          const float* __restrict__ beta,
                          void* __restrict__ outv, int ldC) {
  __shared__ __align__(16) bf16 As[2][64][LDS_STRIDE];
  __shared__ __align__(16) bf16 Bs[2][64][LDS_STRIDE];

  const int tid  = threadIdx.x;
  const int m0   = blockIdx.y * 64;
  const int n0   = blockIdx.x * 64;
  const int wave = tid >> 5, lane = tid & 31;
  const int hl   = lane >> 4, l16 = lane & 15;
  const int mrow = (wave >> 1) * 32, ncol = (wave & 1) * 32;
  // each thread async-copies 2x16B for A and 2x16B for B per tile
  const int r0 = (tid * 2) >> 2, q0 = (tid * 2) & 3;      // chunk tid*2
  const int r1 = (tid * 2 + 1) >> 2, q1 = (tid * 2 + 1) & 3;

  v8f acc[2][2] = {};

#define STAGE(buf, kk)                                                          \
  do {                                                                          \
    async_cp16(A + (size_t)(m0 + r0) * ldA + (kk) + q0 * 8,                     \
               lds_off_of(&As[buf][r0][q0 * 8]));                               \
    async_cp16(Wp + (size_t)(n0 + r0) * K + (kk) + q0 * 8,                      \
               lds_off_of(&Bs[buf][r0][q0 * 8]));                               \
    async_cp16(A + (size_t)(m0 + r1) * ldA + (kk) + q1 * 8,                     \
               lds_off_of(&As[buf][r1][q1 * 8]));                               \
    async_cp16(Wp + (size_t)(n0 + r1) * K + (kk) + q1 * 8,                      \
               lds_off_of(&Bs[buf][r1][q1 * 8]));                               \
  } while (0)

  STAGE(0, 0);
  async_wait0();
  __syncthreads();

  int cur = 0;
  for (int k0 = 0; k0 < K; k0 += 32) {
    const bool more = (k0 + 32) < K;
    if (more) {
      if (cur) STAGE(0, k0 + 32); else STAGE(1, k0 + 32);
    }
    Frag fa[2], fb[2];
#pragma unroll
    for (int mi = 0; mi < 2; ++mi) {
      const bf16* pr = &As[cur][mrow + mi * 16 + l16][0];
      fa[mi].h[0] = *(const v8bf*)(pr + hl * 8);        // K 0-7 / 8-15
      fa[mi].h[1] = *(const v8bf*)(pr + 16 + hl * 8);   // K 16-23 / 24-31
    }
#pragma unroll
    for (int ni = 0; ni < 2; ++ni) {
      const bf16* pr = &Bs[cur][ncol + ni * 16 + l16][0];
      fb[ni].h[0] = *(const v8bf*)(pr + hl * 8);
      fb[ni].h[1] = *(const v8bf*)(pr + 16 + hl * 8);
    }
#pragma unroll
    for (int mi = 0; mi < 2; ++mi)
#pragma unroll
      for (int ni = 0; ni < 2; ++ni)
        acc[mi][ni] = __builtin_amdgcn_wmma_f32_16x16x32_bf16(
            false, fa[mi].v, false, fb[ni].v, (short)0, acc[mi][ni], false, false);
    if (more) { async_wait0(); __syncthreads(); }
    cur ^= 1;
  }
#undef STAGE

  // epilogue: D lane holds n = lane&15, m = r + 8*half
#pragma unroll
  for (int ni = 0; ni < 2; ++ni) {
    const int n = n0 + ncol + ni * 16 + l16;
    const float bs = HAS_BIAS ? bias[n] : 0.f;
    const float gm = HAS_GB ? gamma[n] : 1.f;
    const float bt = HAS_GB ? beta[n] : 0.f;
#pragma unroll
    for (int mi = 0; mi < 2; ++mi) {
#pragma unroll
      for (int r = 0; r < 8; ++r) {
        const int m = m0 + mrow + mi * 16 + hl * 8 + r;
        float v = acc[mi][ni][r] + bs;
        if (HAS_GB) v = v * gm + bt;
        if (SILU) v = silu_f(v);
        if (OUT_NCHW) {
          const int b = m >> 12, p = m & 4095;
          ((float*)outv)[((size_t)(b * ldC + n)) * 4096 + p] = v;
        } else {
          ((bf16*)outv)[(size_t)m * ldC + n] = (bf16)v;
        }
      }
    }
  }
}

// ---------------------------------------------------------------------------
// Direct 3x3 SAME conv via WMMA. A: channels-last bf16. Wp: packed [9][N][K]
// bf16. One 64-pixel M tile == one image row (W=64) -> trivial halo.
// ---------------------------------------------------------------------------
template <bool SILU, bool RS>
__global__ void conv3_wmma(const bf16* __restrict__ A, int ldA,
                           const bf16* __restrict__ Wp, int K, int Ntot,
                           const float* __restrict__ gamma,
                           const float* __restrict__ beta,
                           const float* __restrict__ rowscale,
                           bf16* __restrict__ out, int ldC) {
  __shared__ __align__(16) bf16 As[64][LDS_STRIDE];
  __shared__ __align__(16) bf16 Bs[64][LDS_STRIDE];

  const int tid  = threadIdx.x;
  const int m0   = blockIdx.y * 64;
  const int n0   = blockIdx.x * 64;
  const int wave = tid >> 5, lane = tid & 31;
  const int hl   = lane >> 4, l16 = lane & 15;
  const int mrow = (wave >> 1) * 32, ncol = (wave & 1) * 32;
  const int row  = (m0 >> 6) & 63;
  const int r    = tid >> 1, hf = tid & 1;   // staging: 16B chunk at half-row start each

  v8f acc[2][2] = {};

  for (int tap = 0; tap < 9; ++tap) {
    const int dy = tap / 3 - 1, dx = tap % 3 - 1;
    const bool rowok = (unsigned)(row + dy) < 64u;
    const int col = r + dx;
    const bool ok = rowok && (unsigned)col < 64u;
    for (int k0 = 0; k0 < K; k0 += 32) {
      uint4 va = {0u, 0u, 0u, 0u};
      if (ok)
        va = *(const uint4*)(A + (size_t)(m0 + r + dx + dy * 64) * ldA + k0 + hf * 16);
      *(uint4*)&As[r][hf * 16] = va;
      *(uint4*)&Bs[r][hf * 16] =
          *(const uint4*)(Wp + ((size_t)tap * Ntot + n0 + r) * K + k0 + hf * 16);
      __syncthreads();

      Frag fa[2], fb[2];
#pragma unroll
      for (int mi = 0; mi < 2; ++mi) {
        const bf16* pr = &As[mrow + mi * 16 + l16][0];
        fa[mi].h[0] = *(const v8bf*)(pr + hl * 8);
        fa[mi].h[1] = *(const v8bf*)(pr + 16 + hl * 8);
      }
#pragma unroll
      for (int ni = 0; ni < 2; ++ni) {
        const bf16* pr = &Bs[ncol + ni * 16 + l16][0];
        fb[ni].h[0] = *(const v8bf*)(pr + hl * 8);
        fb[ni].h[1] = *(const v8bf*)(pr + 16 + hl * 8);
      }
#pragma unroll
      for (int mi = 0; mi < 2; ++mi)
#pragma unroll
        for (int ni = 0; ni < 2; ++ni)
          acc[mi][ni] = __builtin_amdgcn_wmma_f32_16x16x32_bf16(
              false, fa[mi].v, false, fb[ni].v, (short)0, acc[mi][ni], false, false);
      __syncthreads();
    }
  }

#pragma unroll
  for (int ni = 0; ni < 2; ++ni) {
    const int n = n0 + ncol + ni * 16 + l16;
    const float gm = gamma[n], bt = beta[n];
#pragma unroll
    for (int mi = 0; mi < 2; ++mi) {
#pragma unroll
      for (int rr = 0; rr < 8; ++rr) {
        const int m = m0 + mrow + mi * 16 + hl * 8 + rr;
        float v = acc[mi][ni][rr] * gm + bt;
        if (SILU) v = silu_f(v);
        if (RS)   v *= rowscale[m];
        out[(size_t)m * ldC + n] = (bf16)v;
      }
    }
  }
}

// ---------------------------------------------------------------------------
// DCNv3 core: depthwise 3x3 + bias + LayerNorm + exact GELU, offset/mask
// projections, softmax over 9 taps, bilinear gather of xp (zero-padded
// grid_sample), mask-weighted sum. One pixel per block, 128 thr = channels.
// ---------------------------------------------------------------------------
__global__ void dcn_core(const bf16* __restrict__ t1,  // [16384,128]
                         const bf16* __restrict__ xp,  // [16384,128]
                         const float* __restrict__ dw_w, const float* __restrict__ dw_b,
                         const float* __restrict__ ln_g, const float* __restrict__ ln_b,
                         const float* __restrict__ off_w, const float* __restrict__ off_b,
                         const float* __restrict__ msk_w, const float* __restrict__ msk_b,
                         bf16* __restrict__ out) {
  const int m = blockIdx.x;
  const int c = threadIdx.x;
  const int b = m >> 12, p = m & 4095, h = p >> 6, w = p & 63;

  __shared__ float x1s[128];
  __shared__ float red[128];
  __shared__ float offs[18];
  __shared__ float msks[9];

  float s = dw_b[c];
#pragma unroll
  for (int ky = 0; ky < 3; ++ky)
#pragma unroll
    for (int kx = 0; kx < 3; ++kx) {
      const int hh = h + ky - 1, ww = w + kx - 1;
      if ((unsigned)hh < 64u && (unsigned)ww < 64u)
        s += (float)t1[((size_t)((b << 12) + (hh << 6) + ww)) * 128 + c] *
             dw_w[(ky * 3 + kx) * 128 + c];
    }

  red[c] = s; __syncthreads();
  for (int st = 64; st > 0; st >>= 1) { if (c < st) red[c] += red[c + st]; __syncthreads(); }
  const float mu = red[0] * (1.f / 128.f); __syncthreads();
  const float d0 = s - mu;
  red[c] = d0 * d0; __syncthreads();
  for (int st = 64; st > 0; st >>= 1) { if (c < st) red[c] += red[c + st]; __syncthreads(); }
  const float var = red[0] * (1.f / 128.f); __syncthreads();
  float x1 = d0 * rsqrtf(var + 1e-5f) * ln_g[c] + ln_b[c];
  x1 = 0.5f * x1 * (1.f + erff(x1 * 0.70710678118654752f));
  x1s[c] = x1; __syncthreads();

  if (c < 18) {
    float a = off_b[c];
    for (int k = 0; k < 128; ++k) a += x1s[k] * off_w[k * 18 + c];
    offs[c] = a;
  } else if (c < 27) {
    const int j = c - 18;
    float a = msk_b[j];
    for (int k = 0; k < 128; ++k) a += x1s[k] * msk_w[k * 9 + j];
    msks[j] = a;
  }
  __syncthreads();
  if (c == 0) {
    float mx = msks[0];
    for (int j = 1; j < 9; ++j) mx = fmaxf(mx, msks[j]);
    float sm = 0.f;
    for (int j = 0; j < 9; ++j) { msks[j] = expf(msks[j] - mx); sm += msks[j]; }
    const float inv = 1.f / sm;
    for (int j = 0; j < 9; ++j) msks[j] *= inv;
  }
  __syncthreads();

  float acc = 0.f;
#pragma unroll
  for (int t = 0; t < 9; ++t) {
    const float px = (float)(w + 1 + (t / 3) - 1) + offs[2 * t];
    const float py = (float)(h + 1 + (t % 3) - 1) + offs[2 * t + 1];
    const float x0f = floorf(px), y0f = floorf(py);
    const int x0 = (int)x0f, y0 = (int)y0f;
    const float fx = px - x0f, fy = py - y0f;
    float v[4];
#pragma unroll
    for (int q = 0; q < 4; ++q) {
      const int xi = x0 + (q & 1), yi = y0 + (q >> 1);
      const bool okq = (xi >= 1) & (xi <= 64) & (yi >= 1) & (yi <= 64);
      v[q] = okq ? (float)xp[((size_t)((b << 12) + ((yi - 1) << 6) + (xi - 1))) * 128 + c]
                 : 0.f;
    }
    const float bl = (v[0] * (1.f - fx) + v[1] * fx) * (1.f - fy) +
                     (v[2] * (1.f - fx) + v[3] * fx) * fy;
    acc += msks[t] * bl;
  }
  out[(size_t)m * 128 + c] = (bf16)acc;
}

// ---------------------------------------------------------------------------
// MaxSigmoid attention: aw[m] = sigmoid(max_n <h2[m,:], g[b,n,:]>/sqrt(128)+bias)
// ---------------------------------------------------------------------------
__global__ void attn_scores(const bf16* __restrict__ h2, int ldA,
                            const bf16* __restrict__ g,   // [B*80,128]
                            const float* __restrict__ bias,
                            float* __restrict__ aw) {
  const int m = blockIdx.x;
  const int t = threadIdx.x;
  const int b = m >> 12;
  __shared__ float xs[128];
  __shared__ float red[128];
  xs[t] = (float)h2[(size_t)m * ldA + t];
  __syncthreads();
  float d = -1e30f;
  if (t < 80) {
    const bf16* gr = g + (size_t)(b * 80 + t) * 128;
    float s = 0.f;
    for (int k = 0; k < 128; ++k) s += xs[k] * (float)gr[k];
    d = s;
  }
  red[t] = d; __syncthreads();
  for (int st = 64; st > 0; st >>= 1) { if (t < st) red[t] = fmaxf(red[t], red[t + st]); __syncthreads(); }
  if (t == 0)
    aw[m] = 1.f / (1.f + expf(-(red[0] * 0.088388347648318447f + bias[0])));
}

// ---------------------------------------------------------------------------
extern "C" void kernel_launch(void* const* d_in, const int* in_sizes, int n_in,
                              void* d_out, int out_size, void* d_ws, size_t ws_size,
                              hipStream_t stream) {
  const float* x           = (const float*)d_in[0];
  const float* guide       = (const float*)d_in[1];
  const float* cv1_w       = (const float*)d_in[2];
  const float* cv1_g       = (const float*)d_in[3];
  const float* cv1_b       = (const float*)d_in[4];
  const float* m_cv1_w     = (const float*)d_in[5];
  const float* m_cv1_g     = (const float*)d_in[6];
  const float* m_cv1_b     = (const float*)d_in[7];
  const float* m_pre_w     = (const float*)d_in[8];
  const float* m_pre_g     = (const float*)d_in[9];
  const float* m_pre_b     = (const float*)d_in[10];
  const float* m_dw_w      = (const float*)d_in[11];
  const float* m_dw_b      = (const float*)d_in[12];
  const float* m_ln_g      = (const float*)d_in[13];
  const float* m_ln_b      = (const float*)d_in[14];
  const float* m_off_w     = (const float*)d_in[15];
  const float* m_off_b     = (const float*)d_in[16];
  const float* m_msk_w     = (const float*)d_in[17];
  const float* m_msk_b     = (const float*)d_in[18];
  const float* m_in_w      = (const float*)d_in[19];
  const float* m_in_b      = (const float*)d_in[20];
  const float* m_out_w     = (const float*)d_in[21];
  const float* m_out_b     = (const float*)d_in[22];
  const float* attn_gl_w   = (const float*)d_in[25];
  const float* attn_gl_b   = (const float*)d_in[26];
  const float* attn_bias   = (const float*)d_in[27];
  const float* attn_proj_w = (const float*)d_in[28];
  const float* attn_proj_g = (const float*)d_in[29];
  const float* attn_proj_b = (const float*)d_in[30];
  const float* m_bn_g      = (const float*)d_in[23];
  const float* m_bn_b      = (const float*)d_in[24];
  const float* cv2_w       = (const float*)d_in[31];
  const float* cv2_g       = (const float*)d_in[32];
  const float* cv2_b       = (const float*)d_in[33];

  const size_t M = 16384;
  // ---- workspace layout ----
  float* aw  = (float*)d_ws;                 // [16384] f32
  bf16* base = (bf16*)(aw + M);
  size_t o = 0;
  bf16* xcl   = base + o; o += M * 256;       // x channels-last bf16
  bf16* ycat  = base + o; o += M * 640;       // concat y0|y1|h1|h2|attn
  bf16* t0    = base + o; o += M * 128;       // conv3 out / dcn samp (reused)
  bf16* t1    = base + o; o += M * 128;       // pre-conv out
  bf16* xp    = base + o; o += M * 128;       // dcn input_proj
  bf16* gub   = base + o; o += 320 * 512;     // guide bf16
  bf16* gbuf  = base + o; o += 320 * 128;     // guide-linear out
  bf16* wcv1  = base + o; o += 256 * 256;
  bf16* wcv2  = base + o; o += 256 * 640;
  bf16* wgl   = base + o; o += 128 * 512;
  bf16* wpre  = base + o; o += 2 * 128 * 128;
  bf16* win   = base + o; o += 2 * 128 * 128;
  bf16* wout  = base + o; o += 2 * 128 * 128;
  bf16* wcv3  = base + o; o += 2 * 9 * 128 * 128;
  bf16* wproj = base + o; o += 9 * 128 * 128;
  bf16* samp  = t0;

  const dim3 blk(128);
  const dim3 g2(2, 256);   // N=128, M=16384
  const dim3 g4(4, 256);   // N=256, M=16384
#define PK(n) dim3(((n) + 255) / 256), dim3(256), 0, stream

  // ---- pack weights + inputs to bf16 ----
  pack_nchw<<<PK(4 * 256 * 4096)>>>(x, xcl, 256);
  pack_copy<<<PK(320 * 512)>>>(guide, gub, 320 * 512);
  pack_copy<<<PK(256 * 256)>>>(cv1_w, wcv1, 256 * 256);
  pack_copy<<<PK(256 * 640)>>>(cv2_w, wcv2, 256 * 640);
  pack_t<<<PK(128 * 512)>>>(attn_gl_w, wgl, 512, 128);
  pack_copy<<<PK(2 * 128 * 128)>>>(m_pre_w, wpre, 2 * 128 * 128);
  for (int i = 0; i < 2; ++i) {
    pack_t<<<PK(128 * 128)>>>(m_in_w + (size_t)i * 16384, win + (size_t)i * 16384, 128, 128);
    pack_t<<<PK(128 * 128)>>>(m_out_w + (size_t)i * 16384, wout + (size_t)i * 16384, 128, 128);
    pack_conv<<<PK(9 * 128 * 128)>>>(m_cv1_w + (size_t)i * 147456,
                                     wcv3 + (size_t)i * 147456, 128, 128);
  }
  pack_conv<<<PK(9 * 128 * 128)>>>(attn_proj_w, wproj, 128, 128);

  // ---- cv1: 1x1 conv 256->256, BN+SiLU -> ycat cols 0..255 ----
  gemm_wmma<false, false, true, true><<<g4, blk, 0, stream>>>(
      xcl, 256, wcv1, 256, nullptr, cv1_g, cv1_b, ycat, 640);

  for (int i = 0; i < 2; ++i) {
    const bf16* src = ycat + 128 + 128 * i;  // y1 (i=0) / h1 (i=1), ld 640
    conv3_wmma<true, false><<<g2, blk, 0, stream>>>(
        src, 640, wcv3 + (size_t)i * 147456, 128, 128,
        m_cv1_g + i * 128, m_cv1_b + i * 128, nullptr, t0, 128);
    gemm_wmma<false, false, true, true><<<g2, blk, 0, stream>>>(
        t0, 128, wpre + (size_t)i * 16384, 128,
        nullptr, m_pre_g + i * 128, m_pre_b + i * 128, t1, 128);
    gemm_wmma<false, true, false, false><<<g2, blk, 0, stream>>>(
        t1, 128, win + (size_t)i * 16384, 128,
        m_in_b + i * 128, nullptr, nullptr, xp, 128);
    dcn_core<<<dim3(16384), blk, 0, stream>>>(
        t1, xp, m_dw_w + (size_t)i * 1152, m_dw_b + i * 128,
        m_ln_g + i * 128, m_ln_b + i * 128,
        m_off_w + (size_t)i * 2304, m_off_b + i * 18,
        m_msk_w + (size_t)i * 1152, m_msk_b + i * 9, samp);
    gemm_wmma<false, true, true, true><<<g2, blk, 0, stream>>>(
        samp, 128, wout + (size_t)i * 16384, 128,
        m_out_b + i * 128, m_bn_g + i * 128, m_bn_b + i * 128,
        ycat + 256 + 128 * i, 640);
  }

  // ---- guide linear: [320,512] @ [512,128] + bias ----
  gemm_wmma<false, true, false, false><<<dim3(2, 5), blk, 0, stream>>>(
      gub, 512, wgl, 512, attn_gl_b, nullptr, nullptr, gbuf, 128);

  // ---- max-sigmoid attention weights over h2 (ycat cols 384..511) ----
  attn_scores<<<dim3(16384), blk, 0, stream>>>(ycat + 384, 640, gbuf, attn_bias, aw);

  // ---- attn proj: 3x3 conv + BN (no act), scaled by aw -> cols 512..639 ----
  conv3_wmma<false, true><<<g2, blk, 0, stream>>>(
      ycat + 384, 640, wproj, 128, 128, attn_proj_g, attn_proj_b, aw,
      ycat + 512, 640);

  // ---- cv2: 1x1 conv 640->256 + BN + SiLU -> fp32 NCHW output ----
  gemm_wmma<true, false, true, true><<<g4, blk, 0, stream>>>(
      ycat, 640, wcv2, 640, nullptr, cv2_g, cv2_b, d_out, 256);

#undef PK
  (void)in_sizes; (void)n_in; (void)out_size; (void)ws_size;
}
